// ScaledDotProductAttention_47674136985727
// MI455X (gfx1250) — compile-verified, matrix operand order checked by reading, not run on hardware
//
#include <hip/hip_runtime.h>

// ---------------------------------------------------------------------------
// Flash attention forward for [4,16,2048,128] fp32 on gfx1250 (MI455X).
// bf16 WMMA (v_wmma_f32_16x16x32_bf16), f32 accumulate, online softmax.
// Software-pipelined: double-buffered LDS K/V tiles, global prefetch into
// registers overlapped with the WMMA block, one barrier per KV step,
// last iteration peeled so the steady-state loop is branch-free.
// Softmax row reductions use DPP16 (VALU) instead of ds_bpermute.
// ---------------------------------------------------------------------------

typedef __attribute__((ext_vector_type(16))) __bf16 v16bf;
typedef __attribute__((ext_vector_type(2)))  __bf16 bf16x2;
typedef __attribute__((ext_vector_type(8)))  float  v8f;

#define S_LEN 2048
#define HD    128
#define BM    128   // query rows per workgroup (8 waves x 16)
#define BN    32    // kv keys per block step
#define NW    8     // waves per workgroup
#define NT    (S_LEN / BN)

struct BFPack { uint4 lo, hi; };           // 32 bytes == 16 bf16
struct BF16V  { unsigned short h[16]; };

// ---- f32 -> bf16 through native casts (ISel picks v_cvt_* if available) ---
static __device__ __forceinline__ unsigned pk2(float a, float b) {
  bf16x2 t;
  t.x = (__bf16)a;
  t.y = (__bf16)b;
  return __builtin_bit_cast(unsigned, t);
}
static __device__ __forceinline__ unsigned short f2bf(float f) {
  __bf16 h = (__bf16)f;
  return __builtin_bit_cast(unsigned short, h);
}

static __device__ __forceinline__ v8f zero8() {
  v8f z;
#pragma unroll
  for (int i = 0; i < 8; ++i) z[i] = 0.0f;
  return z;
}

// ---- DPP16 row(16-lane) reductions (xor1/2/4/8 equivalents) ---------------
template <int CTRL>
static __device__ __forceinline__ float dppmov(float x) {
  return __builtin_bit_cast(
      float, __builtin_amdgcn_update_dpp(
                 0, __builtin_bit_cast(int, x), CTRL, 0xf, 0xf, true));
}
static __device__ __forceinline__ float rmax16(float x) {
  x = fmaxf(x, dppmov<0xB1>(x));   // quad_perm [1,0,3,2]  == xor 1
  x = fmaxf(x, dppmov<0x4E>(x));   // quad_perm [2,3,0,1]  == xor 2
  x = fmaxf(x, dppmov<0x141>(x));  // row_half_mirror      == xor 4 (post-quad)
  x = fmaxf(x, dppmov<0x140>(x));  // row_mirror           == xor 8 (post-half)
  return x;
}
static __device__ __forceinline__ float rsum16(float x) {
  x += dppmov<0xB1>(x);
  x += dppmov<0x4E>(x);
  x += dppmov<0x141>(x);
  x += dppmov<0x140>(x);
  return x;
}

static __device__ __forceinline__ float f4get(const float4& v, int i) {
  switch (i) { case 0: return v.x; case 1: return v.y; case 2: return v.z; }
  return v.w;
}

__global__ __launch_bounds__(256)
void fa_fwd(const float* __restrict__ Q, const float* __restrict__ K,
            const float* __restrict__ V, float* __restrict__ O) {
  // Double-buffered tiles:
  //   K: bf16 pairs [buf][key][dim/2]    2 x 8 KB
  //   V: bf16 transposed [buf][dim][key] 2 x 8 KB
  //   P: per-wave scratch [wave][m][key]     8 KB   => 40 KB total
  __shared__ unsigned       ktile[2][BN][HD / 2];
  __shared__ unsigned short vtile[2][HD][BN];
  __shared__ unsigned short ptile[NW][16][BN];

  const int tid  = threadIdx.x;
  const int wave = tid >> 5;
  const int lane = tid & 31;
  const int lrow = lane & 15;   // M (or N) index within a 16-wide tile
  const int hf   = lane >> 4;   // which 16-lane half

  const int    bh   = blockIdx.y;
  const size_t base = (size_t)bh * (size_t)(S_LEN * HD);
  const float* Qb = Q + base;
  const float* Kb = K + base;
  const float* Vb = V + base;
  float*       Ob = O + base;

  const int m0 = blockIdx.x * BM + wave * 16;

  // Staging thread mappings
  const int kkey = tid >> 3;           // 0..31   (K: one key row)
  const int kseg = (tid & 7) * 16;     // 0..112  (K: 16-dim segment)
  const int vkb  = (tid >> 5) * 4;     // 0..28   (V: 4-key block)
  const int vdb  = (tid & 31) * 4;     // 0..124  (V: 4-dim block)

  // ---- Load Q tile (16x128) straight into WMMA A-operand layout ----------
  // A layout (16-bit, 16x32): lanes 0-15 hold M=lane, element j maps to
  // k = 16*(j>>3) + 8*half + (j&7).  Fold 1/sqrt(128)*log2(e) into Q.
  const float qscale = 0.088388347648318447f * 1.4426950408889634f;
  v16bf qa[4];
  {
    const float* qrow = Qb + (size_t)(m0 + lrow) * HD;
#pragma unroll
    for (int c = 0; c < 4; ++c) {
      BF16V t;
#pragma unroll
      for (int g = 0; g < 2; ++g) {
        const float4* qp = (const float4*)(qrow + c * 32 + g * 16 + hf * 8);
        float4 f0 = qp[0], f1 = qp[1];
        t.h[g * 8 + 0] = f2bf(f0.x * qscale);
        t.h[g * 8 + 1] = f2bf(f0.y * qscale);
        t.h[g * 8 + 2] = f2bf(f0.z * qscale);
        t.h[g * 8 + 3] = f2bf(f0.w * qscale);
        t.h[g * 8 + 4] = f2bf(f1.x * qscale);
        t.h[g * 8 + 5] = f2bf(f1.y * qscale);
        t.h[g * 8 + 6] = f2bf(f1.z * qscale);
        t.h[g * 8 + 7] = f2bf(f1.w * qscale);
      }
      qa[c] = __builtin_bit_cast(v16bf, t);
    }
  }

  // ---- Online-softmax state (C layout: lane owns rows M = r + 8*hf) ------
  v8f acc[8];
#pragma unroll
  for (int i = 0; i < 8; ++i) acc[i] = zero8();
  float mrow[8], lsum[8];
#pragma unroll
  for (int r = 0; r < 8; ++r) { mrow[r] = -3.0e38f; lsum[r] = 0.0f; }

  // Prefetch registers (tile i+1 while computing tile i)
  float4 kf[4], vf[4];

  auto prefetch = [&](int kv0) {
    const float4* kp = (const float4*)(Kb + (size_t)(kv0 + kkey) * HD + kseg);
#pragma unroll
    for (int i = 0; i < 4; ++i) kf[i] = kp[i];
#pragma unroll
    for (int i = 0; i < 4; ++i)
      vf[i] = *(const float4*)(Vb + (size_t)(kv0 + vkb + i) * HD + vdb);
  };

  auto stage = [&](int nb) {
    uint4 a = make_uint4(pk2(kf[0].x, kf[0].y), pk2(kf[0].z, kf[0].w),
                         pk2(kf[1].x, kf[1].y), pk2(kf[1].z, kf[1].w));
    uint4 b = make_uint4(pk2(kf[2].x, kf[2].y), pk2(kf[2].z, kf[2].w),
                         pk2(kf[3].x, kf[3].y), pk2(kf[3].z, kf[3].w));
    *(uint4*)&ktile[nb][kkey][kseg / 2]     = a;
    *(uint4*)&ktile[nb][kkey][kseg / 2 + 4] = b;
#pragma unroll
    for (int d = 0; d < 4; ++d) {
      *(uint2*)&vtile[nb][vdb + d][vkb] =
          make_uint2(pk2(f4get(vf[0], d), f4get(vf[1], d)),
                     pk2(f4get(vf[2], d), f4get(vf[3], d)));
    }
  };

  auto compute_tile = [&](int buf) {
    // ---- S = Q * K^T  (2 column tiles x 4 k-steps) ----
    v8f s[2];
#pragma unroll
    for (int t = 0; t < 2; ++t) {
      s[t] = zero8();
      const int n = t * 16 + lrow;       // key index of this lane's B column
#pragma unroll
      for (int c = 0; c < 4; ++c) {
        BFPack kpk;
        kpk.lo = *(const uint4*)&ktile[buf][n][c * 16 + hf * 4];
        kpk.hi = *(const uint4*)&ktile[buf][n][c * 16 + 8 + hf * 4];
        v16bf kb = __builtin_bit_cast(v16bf, kpk);
        s[t] = __builtin_amdgcn_wmma_f32_16x16x32_bf16(
            false, qa[c], false, kb, (short)0, s[t], false, false);
      }
    }

    // ---- Online softmax; store P rows into wave-private LDS as we go ----
    v8f sclv;
#pragma unroll
    for (int r = 0; r < 8; ++r) {
      float mx   = rmax16(fmaxf(s[0][r], s[1][r]));
      float mnew = fmaxf(mrow[r], mx);
      float sc   = __builtin_amdgcn_exp2f(mrow[r] - mnew);
      mrow[r] = mnew;
      sclv[r] = sc;
      float p0 = __builtin_amdgcn_exp2f(s[0][r] - mnew);
      float p1 = __builtin_amdgcn_exp2f(s[1][r] - mnew);
      lsum[r] = lsum[r] * sc + rsum16(p0 + p1);
      unsigned pw = pk2(p0, p1);         // one packed convert
      ptile[wave][r + 8 * hf][lrow]      = (unsigned short)(pw & 0xffffu);
      ptile[wave][r + 8 * hf][16 + lrow] = (unsigned short)(pw >> 16);
    }
#pragma unroll
    for (int dc = 0; dc < 8; ++dc) acc[dc] *= sclv;   // v_pk_mul_f32-able

    // ---- C->A relayout of P (LDS ops are in-order within a wave) ----
    BFPack pp;
    pp.lo = *(const uint4*)&ptile[wave][lrow][8 * hf];
    pp.hi = *(const uint4*)&ptile[wave][lrow][16 + 8 * hf];
    v16bf pa = __builtin_bit_cast(v16bf, pp);

    // ---- O += P * V  (8 D-chunks of 16) ----
#pragma unroll
    for (int dc = 0; dc < 8; ++dc) {
      const int n = dc * 16 + lrow;      // output dim of this lane's B column
      BFPack vpk;
      vpk.lo = *(const uint4*)&vtile[buf][n][8 * hf];
      vpk.hi = *(const uint4*)&vtile[buf][n][16 + 8 * hf];
      v16bf vb = __builtin_bit_cast(v16bf, vpk);
      acc[dc] = __builtin_amdgcn_wmma_f32_16x16x32_bf16(
          false, pa, false, vb, (short)0, acc[dc], false, false);
    }
  };

  // ---- Prologue: fetch + stage tile 0 into buffer 0 ----------------------
  prefetch(0);
  stage(0);
  __syncthreads();

  // ---- Steady-state loop: branch-free prefetch/compute/stage -------------
  for (int it = 0; it < NT - 1; ++it) {
    const int buf = it & 1;
    prefetch((it + 1) * BN);   // VMEM latency hidden under the WMMAs
    compute_tile(buf);
    stage(buf ^ 1);
    __syncthreads();
  }

  // ---- Peeled final tile --------------------------------------------------
  compute_tile((NT - 1) & 1);

  // ---- Normalize and store -----------------------------------------------
#pragma unroll
  for (int r = 0; r < 8; ++r) {
    const float inv = 1.0f / lsum[r];
    float* orow = Ob + (size_t)(m0 + r + 8 * hf) * HD;
#pragma unroll
    for (int dc = 0; dc < 8; ++dc) orow[dc * 16 + lrow] = acc[dc][r] * inv;
  }
}

extern "C" void kernel_launch(void* const* d_in, const int* in_sizes, int n_in,
                              void* d_out, int out_size, void* d_ws, size_t ws_size,
                              hipStream_t stream) {
  const float* q = (const float*)d_in[0];
  const float* k = (const float*)d_in[1];
  const float* v = (const float*)d_in[2];
  float*       o = (float*)d_out;
  const int bh = in_sizes[0] / (S_LEN * HD);   // batch * heads = 64
  dim3 grid(S_LEN / BM, bh);
  fa_fwd<<<grid, 256, 0, stream>>>(q, k, v, o);
}
